// AttnDecoderRNN_68917045232366
// MI455X (gfx1250) — compile-verified, matrix operand order checked by reading, not run on hardware
//
#include <hip/hip_runtime.h>
#include <hip/hip_bf16.h>
#include <math.h>

// ---------------- problem constants ----------------
#define BB 1024
#define SS 100
#define HH 256
#define EE 128
#define VV 50000

typedef __attribute__((ext_vector_type(16))) __bf16 v16bf;
typedef __attribute__((ext_vector_type(8)))  float  v8f;

// ---------------- helpers ----------------

__device__ __forceinline__ unsigned short f2bf(float f) {
  return (unsigned short)(__float_as_uint(f) >> 16);  // truncating f32 -> bf16
}

// Load one 16x32 bf16 fragment slice for this lane from a pre-converted bf16
// matrix. p (ushort units) points at the first of 8 contiguous bf16
// (k = kb + 8*(lane>>4)); the second run of 8 is at p+16 elements.
// Two 16-byte loads, zero conversion ALU.
__device__ __forceinline__ v16bf load_frag(const unsigned short* __restrict__ p) {
  union { v16bf v; uint4 q[2]; } u;
  u.q[0] = *reinterpret_cast<const uint4*>(p);
  u.q[1] = *reinterpret_cast<const uint4*>(p + 16);
  return u.v;
}

// Block reductions for 256-thread 1D blocks (8 wave32 waves).
__device__ __forceinline__ float blockReduceSum(float v, float* red) {
#pragma unroll
  for (int off = 16; off > 0; off >>= 1) v += __shfl_xor(v, off, 32);
  int t = threadIdx.x;
  if ((t & 31) == 0) red[t >> 5] = v;
  __syncthreads();
  if (t == 0) {
    float s = 0.f;
#pragma unroll
    for (int i = 0; i < 8; ++i) s += red[i];
    red[8] = s;
  }
  __syncthreads();
  float out = red[8];
  __syncthreads();
  return out;
}

__device__ __forceinline__ float blockReduceMax(float v, float* red) {
#pragma unroll
  for (int off = 16; off > 0; off >>= 1) v = fmaxf(v, __shfl_xor(v, off, 32));
  int t = threadIdx.x;
  if ((t & 31) == 0) red[t >> 5] = v;
  __syncthreads();
  if (t == 0) {
    float m = red[0];
#pragma unroll
    for (int i = 1; i < 8; ++i) m = fmaxf(m, red[i]);
    red[8] = m;
  }
  __syncthreads();
  float out = red[8];
  __syncthreads();
  return out;
}

// ---------------- kernels ----------------

// Bulk f32 -> bf16 conversion, 2 elements per thread (one packed dword out).
__global__ void f32_to_bf16_kernel(const float* __restrict__ in,
                                   unsigned* __restrict__ out, int n2) {
  int i = blockIdx.x * blockDim.x + threadIdx.x;
  if (i < n2) {
    float2 f = reinterpret_cast<const float2*>(in)[i];
    out[i] = (__float_as_uint(f.x) >> 16) | (__float_as_uint(f.y) & 0xFFFF0000u);
  }
}

// x[b,:] = emb[token[b],:]  (writes f32 copy for p_gen and bf16 copy for GEMM)
__global__ void embed_kernel(const int* __restrict__ tok,
                             const float* __restrict__ emb,
                             float* __restrict__ x,
                             unsigned short* __restrict__ xb) {
  int b = blockIdx.x;
  int e = threadIdx.x;  // blockDim.x == EE
  float v = emb[(size_t)tok[b] * EE + e];
  x[(size_t)b * EE + e]  = v;
  xb[(size_t)b * EE + e] = f2bf(v);
}

// C(MxN) = A(MxK) * W(NxK)^T + bias(N); A,W pre-converted bf16, C f32.
// Each wave computes a 64(M) x 16(N) strip: 4 accumulators sharing one B
// fragment per K-step (4x B reuse). block = (32,4): 4 waves stacked in M.
// grid = (N/16, M/256). Compile-time K -> fully unrolled K loop.
template <int K>
__global__ void wmma_gemm_bf16(const unsigned short* __restrict__ A,
                               const unsigned short* __restrict__ W,
                               const float* __restrict__ bias,
                               float* __restrict__ C,
                               int N) {
  int nBase = blockIdx.x * 16;
  int mBase = blockIdx.y * 256 + threadIdx.y * 64;  // wave's first M row
  int lane  = threadIdx.x;                          // 0..31
  int half  = lane >> 4;                            // 0 or 1
  int c     = lane & 15;

  const unsigned short* wrow = W + (size_t)(nBase + c) * K + 8 * half;
  const unsigned short* arow = A + (size_t)(mBase + c) * K + 8 * half;
  const size_t astep = (size_t)16 * K;              // 16 M rows

  v8f acc0 = {}, acc1 = {}, acc2 = {}, acc3 = {};
#pragma unroll
  for (int kb = 0; kb < K; kb += 32) {
    __builtin_prefetch((const void*)(wrow + kb + 128), 0, 1);  // global_prefetch_b8
    v16bf bf = load_frag(wrow + kb);
    v16bf a0 = load_frag(arow + kb);
    v16bf a1 = load_frag(arow + astep + kb);
    v16bf a2 = load_frag(arow + 2 * astep + kb);
    v16bf a3 = load_frag(arow + 3 * astep + kb);
    acc0 = __builtin_amdgcn_wmma_f32_16x16x32_bf16(false, a0, false, bf, (short)0, acc0, false, false);
    acc1 = __builtin_amdgcn_wmma_f32_16x16x32_bf16(false, a1, false, bf, (short)0, acc1, false, false);
    acc2 = __builtin_amdgcn_wmma_f32_16x16x32_bf16(false, a2, false, bf, (short)0, acc2, false, false);
    acc3 = __builtin_amdgcn_wmma_f32_16x16x32_bf16(false, a3, false, bf, (short)0, acc3, false, false);
  }

  float bv = bias[nBase + c];
  v8f accs[4] = {acc0, acc1, acc2, acc3};
#pragma unroll
  for (int j = 0; j < 4; ++j) {
    int mrow = mBase + 16 * j + (half << 3);  // +8 for lanes 16..31
    size_t base = (size_t)mrow * (size_t)N + (size_t)(nBase + c);
#pragma unroll
    for (int i = 0; i < 8; ++i)
      C[base + (size_t)i * (size_t)N] = accs[j][i] + bv;
  }
}

// GRU gate math: r,z,n from gi/gh (biases already added in the GEMMs).
__global__ void gru_gate_kernel(const float* __restrict__ gi,
                                const float* __restrict__ gh,
                                const float* __restrict__ hprev,
                                float* __restrict__ hnew_out,        // d_out dh region
                                float* __restrict__ dec_ctx,         // ws f32 (B x 512)
                                unsigned short* __restrict__ decb) { // ws bf16 (B x 512)
  int idx = blockIdx.x * blockDim.x + threadIdx.x;  // B*H threads
  int b = idx >> 8, h = idx & 255;
  const float* gib = gi + (size_t)b * 768;
  const float* ghb = gh + (size_t)b * 768;
  float r = 1.f / (1.f + expf(-(gib[h]       + ghb[h])));
  float z = 1.f / (1.f + expf(-(gib[256 + h] + ghb[256 + h])));
  float n = tanhf(gib[512 + h] + r * ghb[512 + h]);
  float hv = (1.f - z) * n + z * hprev[idx];
  hnew_out[idx] = hv;
  dec_ctx[(size_t)b * 512 + h] = hv;
  decb[(size_t)b * 512 + h]    = f2bf(hv);
}

// Per batch row: additive attention scores, softmax over S, context vector.
__global__ void attention_kernel(const float* __restrict__ enc,
                                 const float* __restrict__ dec_ctx,  // h_new cols 0..255
                                 const float* __restrict__ w_h,
                                 const float* __restrict__ w_s,
                                 const float* __restrict__ att_bias,
                                 const float* __restrict__ attn_v,
                                 float* __restrict__ dec_ctx_ctx,    // ws f32 cols 256..511
                                 unsigned short* __restrict__ decb,  // ws bf16 cols 256..511
                                 float* __restrict__ att_out) {      // d_out att region
  __shared__ float red[16];
  __shared__ float sc[SS + 12];
  int b = blockIdx.x;
  int t = threadIdx.x;  // 0..255, one h channel per thread
  const float* encb = enc + (size_t)b * SS * HH;
  float hn = dec_ctx[(size_t)b * 512 + t];
  float q  = w_s[t] * hn + att_bias[0];
  float wv = w_h[t];
  float av = attn_v[t];

  for (int s = 0; s < SS; ++s) {
    float val = tanhf(wv * encb[(size_t)s * HH + t] + q) * av;
    float tot = blockReduceSum(val, red);
    if (t == 0) sc[s] = tot;
  }
  __syncthreads();

  float lv = (t < SS) ? sc[t] : -INFINITY;
  float mx = blockReduceMax(lv, red);
  float e  = (t < SS) ? expf(sc[t] - mx) : 0.f;
  float sm = blockReduceSum(e, red);
  float inv = 1.f / sm;
  if (t < SS) {
    float p = e * inv;
    sc[t] = p;
    att_out[(size_t)b * SS + t] = p;
  }
  __syncthreads();

  float acc = 0.f;
  for (int s = 0; s < SS; ++s) acc += sc[s] * encb[(size_t)s * HH + t];
  dec_ctx_ctx[(size_t)b * 512 + 256 + t] = acc;
  decb[(size_t)b * 512 + 256 + t]        = f2bf(acc);
}

// p_gen[b] = sigmoid( dot(concat(dec_ctx, x), gen_W) + gen_b )
__global__ void pgen_kernel(const float* __restrict__ dec_ctx,
                            const float* __restrict__ x,
                            const float* __restrict__ gen_W,
                            const float* __restrict__ gen_b,
                            float* __restrict__ pg_out) {
  __shared__ float red[16];
  int b = blockIdx.x, t = threadIdx.x;
  float v = 0.f;
  for (int i = t; i < 2 * HH + EE; i += 256) {
    float a = (i < 2 * HH) ? dec_ctx[(size_t)b * 512 + i]
                           : x[(size_t)b * EE + (i - 2 * HH)];
    v += a * gen_W[i];
  }
  float s = blockReduceSum(v, red);
  if (t == 0) pg_out[b] = 1.f / (1.f + expf(-(s + gen_b[0])));
}

// In-place softmax over V per batch row (pv already holds logits+bias).
__global__ void softmaxv_kernel(float* __restrict__ pv) {
  __shared__ float red[16];
  int b = blockIdx.x, t = threadIdx.x;
  float* row = pv + (size_t)b * VV;
  float m = -INFINITY;
  for (int i = t; i < VV; i += 256) m = fmaxf(m, row[i]);
  m = blockReduceMax(m, red);
  float s = 0.f;
  for (int i = t; i < VV; i += 256) {
    float e = expf(row[i] - m);
    row[i] = e;
    s += e;
  }
  s = blockReduceSum(s, red);
  float inv = 1.f / s;
  for (int i = t; i < VV; i += 256) row[i] *= inv;
}

// ---------------- launcher ----------------

static inline void conv_bf16(const float* in, unsigned short* out, size_t n,
                             hipStream_t stream) {
  int n2 = (int)(n / 2);
  f32_to_bf16_kernel<<<(n2 + 255) / 256, 256, 0, stream>>>(
      in, reinterpret_cast<unsigned*>(out), n2);
}

extern "C" void kernel_launch(void* const* d_in, const int* in_sizes, int n_in,
                              void* d_out, int out_size, void* d_ws, size_t ws_size,
                              hipStream_t stream) {
  (void)in_sizes; (void)n_in; (void)out_size; (void)ws_size;

  const int*   tok      = (const int*)  d_in[0];
  const float* hprev    = (const float*)d_in[1];
  const float* enc      = (const float*)d_in[2];
  const float* emb      = (const float*)d_in[3];
  const float* W_ih     = (const float*)d_in[4];
  const float* W_hh     = (const float*)d_in[5];
  const float* b_ih     = (const float*)d_in[6];
  const float* b_hh     = (const float*)d_in[7];
  const float* w_h      = (const float*)d_in[8];
  const float* w_s      = (const float*)d_in[9];
  const float* att_bias = (const float*)d_in[10];
  const float* attn_v   = (const float*)d_in[11];
  const float* gen_W    = (const float*)d_in[12];
  const float* gen_b    = (const float*)d_in[13];
  const float* outh_W   = (const float*)d_in[14];
  const float* outh_b   = (const float*)d_in[15];
  const float* outv_W   = (const float*)d_in[16];
  const float* outv_b   = (const float*)d_in[17];

  // output layout: [dh (B*H)] [p_gen (B)] [p_vocab (B*V)] [att_dist (B*S)]
  float* out   = (float*)d_out;
  float* o_dh  = out;
  float* o_pg  = out + (size_t)BB * HH;
  float* o_pv  = o_pg + BB;
  float* o_att = o_pv + (size_t)BB * VV;

  // ---- workspace layout ----
  // f32 region
  float* ws      = (float*)d_ws;
  float* ws_x    = ws;                                   // B*E
  float* ws_gi   = ws_x   + (size_t)BB * EE;             // B*768
  float* ws_gh   = ws_gi  + (size_t)BB * 768;            // B*768
  float* ws_dec  = ws_gh  + (size_t)BB * 768;            // B*512 [h_new | context]
  float* ws_outh = ws_dec + (size_t)BB * 512;            // B*256
  // bf16 region (ushort units), 16B-aligned (all segment sizes % 8 == 0)
  unsigned short* usb      = (unsigned short*)(ws_outh + (size_t)BB * HH);
  unsigned short* wsb_x    = usb;                              // B*E
  unsigned short* wsb_h    = wsb_x    + (size_t)BB * EE;       // B*H
  unsigned short* wsb_dec  = wsb_h    + (size_t)BB * HH;       // B*512
  unsigned short* wsb_outh = wsb_dec  + (size_t)BB * 512;      // B*256
  unsigned short* wsb_Wih  = wsb_outh + (size_t)BB * HH;       // 768*128
  unsigned short* wsb_Whh  = wsb_Wih  + (size_t)768 * EE;      // 768*256
  unsigned short* wsb_outW = wsb_Whh  + (size_t)768 * HH;      // 256*512
  unsigned short* wsb_outv = wsb_outW + (size_t)HH * 2 * HH;   // 50000*256

  dim3 gblk(32, 4);  // 4 waves, each a 64Mx16N strip -> block = 256M x 16N

  // 0) one-time (per call) f32 -> bf16 operand conversion
  conv_bf16(W_ih,   wsb_Wih,  (size_t)768 * EE,     stream);
  conv_bf16(W_hh,   wsb_Whh,  (size_t)768 * HH,     stream);
  conv_bf16(outh_W, wsb_outW, (size_t)HH * 2 * HH,  stream);
  conv_bf16(outv_W, wsb_outv, (size_t)VV * HH,      stream);
  conv_bf16(hprev,  wsb_h,    (size_t)BB * HH,      stream);

  // 1) embedding gather (f32 + bf16)
  embed_kernel<<<BB, EE, 0, stream>>>(tok, emb, ws_x, wsb_x);

  // 2) gi = x @ W_ih^T + b_ih   (M=1024, N=768, K=128)
  wmma_gemm_bf16<EE><<<dim3(768 / 16, BB / 256), gblk, 0, stream>>>(
      wsb_x, wsb_Wih, b_ih, ws_gi, 768);

  // 3) gh = h @ W_hh^T + b_hh   (M=1024, N=768, K=256)
  wmma_gemm_bf16<HH><<<dim3(768 / 16, BB / 256), gblk, 0, stream>>>(
      wsb_h, wsb_Whh, b_hh, ws_gh, 768);

  // 4) GRU gates -> h_new (d_out dh region + dec_ctx f32/bf16 cols 0..255)
  gru_gate_kernel<<<(BB * HH) / 256, 256, 0, stream>>>(
      ws_gi, ws_gh, hprev, o_dh, ws_dec, wsb_dec);

  // 5) attention: scores, softmax over S, context -> dec_ctx cols 256..511
  attention_kernel<<<BB, 256, 0, stream>>>(
      enc, ws_dec, w_h, w_s, att_bias, attn_v, ws_dec, wsb_dec, o_att);

  // 6) out_h = dec_ctx @ outh_W^T + outh_b  (M=1024, N=256, K=512)
  wmma_gemm_bf16<2 * HH><<<dim3(HH / 16, BB / 256), gblk, 0, stream>>>(
      wsb_dec, wsb_outW, outh_b, ws_outh, HH);
  conv_bf16(ws_outh, wsb_outh, (size_t)BB * HH, stream);

  // 7) p_gen
  pgen_kernel<<<BB, 256, 0, stream>>>(ws_dec, ws_x, gen_W, gen_b, o_pg);

  // 8) logits = out_h @ outv_W^T + outv_b  (M=1024, N=50000, K=256) -> p_vocab
  wmma_gemm_bf16<HH><<<dim3(VV / 16, BB / 256), gblk, 0, stream>>>(
      wsb_outh, wsb_outv, outv_b, o_pv, VV);

  // 9) softmax over V, in place
  softmaxv_kernel<<<BB, 256, 0, stream>>>(o_pv);
}
